// MultiHeadSelfAttention_5403068858378
// MI455X (gfx1250) — matrix operand order repaired and checked by reading, compiled, not run on hardware
//
#include <hip/hip_runtime.h>
#include <hip/hip_bf16.h>

typedef __attribute__((ext_vector_type(16))) _Float16 v16h;
typedef __attribute__((ext_vector_type(8)))  _Float16 v8h;
typedef __attribute__((ext_vector_type(8)))  float    v8f;
typedef unsigned int uint;

static constexpr int Bsz = 16;
static constexpr int Sq  = 1024;
static constexpr int Dm  = 768;
static constexpr int Hh  = 12;
static constexpr int HDd = 64;   // head dim

union U32H2 { uint u; _Float16 h[2]; };

static __device__ inline v8f wmma_f16(v16h a, v16h b, v8f c) {
  return __builtin_amdgcn_wmma_f32_16x16x32_f16(false, a, false, b, (short)0, c,
                                                false, false);
}

// one DS op instead of an address-setup + ds_bpermute pair
static __device__ inline float xor16(float v) {
#if __has_builtin(__builtin_amdgcn_ds_swizzle)
  // group-of-32 swizzle: xor_mask=0x10, and_mask=0x1f  (SWAPX16)
  return __builtin_bit_cast(
      float, __builtin_amdgcn_ds_swizzle(__builtin_bit_cast(int, v), 0x401f));
#else
  return __shfl_xor(v, 16, 32);
#endif
}

// A-fragment (16x32 f16, M x K): lane m = lane&15,
// K(h) = kbase + (h<8 ? h : h+8) + 8*laneHi  -> two contiguous 16B chunks.
static __device__ inline v16h frag_a(const _Float16* base, int ldk, int m,
                                     int kbase, int laneHi) {
  const _Float16* p = base + (long)m * ldk + kbase + 8 * laneHi;
  v16h a;
#pragma unroll
  for (int i = 0; i < 8; ++i) a[i] = p[i];
#pragma unroll
  for (int i = 0; i < 8; ++i) a[8 + i] = p[16 + i];
  return a;
}

// B-fragment (32x16 f16, K x N) from [n][ldk] storage: n = lane&15,
// k(h) = kbase + (h<8 ? h : h+8) + 8*laneHi -> two contiguous 16B chunks.
static __device__ inline v16h frag_b(const _Float16* base, int ldk, int n,
                                     int kbase, int laneHi) {
  const _Float16* p = base + (long)n * ldk + kbase + 8 * laneHi;
  v16h b;
#pragma unroll
  for (int i = 0; i < 8; ++i) b[i] = p[i];
#pragma unroll
  for (int i = 0; i < 8; ++i) b[8 + i] = p[16 + i];
  return b;
}

// ---------------------------------------------------------------- convert
__global__ void cvt_f32_f16_kernel(const float* __restrict__ src,
                                   _Float16* __restrict__ dst, long n) {
  long i = ((long)blockIdx.x * blockDim.x + threadIdx.x) * 4;
  if (i + 3 < n) {
    float4 v = *(const float4*)(src + i);
    dst[i + 0] = (_Float16)v.x;
    dst[i + 1] = (_Float16)v.y;
    dst[i + 2] = (_Float16)v.z;
    dst[i + 3] = (_Float16)v.w;
  } else {
    for (; i < n; ++i) dst[i] = (_Float16)src[i];
  }
}

// ---------------------------------------------------------------- GEMM
// out[m,n] = sum_k A[m,k] * W[n,k] + bias[n]
// Block tile 128x128, 8 waves (4 along M x 2 along N), wave tile 32x64.
// MODE 0: N=2304 QKV projection, scatter to q/k/v [B,H,S,64] f16 (q scaled)
// MODE 1: N=768 output projection, write f32 d_out
template <int MODE>
__launch_bounds__(256)
__global__ void gemm_wmma_kernel(const _Float16* __restrict__ A,
                                 const _Float16* __restrict__ W,
                                 const float* __restrict__ bias, int N, int K,
                                 _Float16* __restrict__ qf,
                                 _Float16* __restrict__ kf,
                                 _Float16* __restrict__ vf,
                                 float* __restrict__ outf) {
  __shared__ __align__(16) _Float16 As[128 * 32];
  __shared__ __align__(16) _Float16 Bs[128 * 32];
  const int tid = threadIdx.x;
  const int lane = tid & 31, laneHi = lane >> 4, lane15 = lane & 15;
  const int wave = tid >> 5;
  const int wm = wave & 3, wn = wave >> 2;       // 4 waves along M, 2 along N
  const long bm = (long)blockIdx.x * 128;
  const long bn = (long)blockIdx.y * 128;

  v8f acc[2][4] = {};

  for (int k0 = 0; k0 < K; k0 += 32) {
    __syncthreads();
    uint* sA = (uint*)As;
    uint* sB = (uint*)Bs;
#pragma unroll
    for (int i = 0; i < 8; ++i) {            // 2048 dwords of A tile
      int idx = tid + i * 256;
      int row = idx >> 4, cu = idx & 15;
      const uint* g = (const uint*)(A + (bm + row) * (long)K + k0);
      sA[idx] = g[cu];
      if (k0 + 32 < K) __builtin_prefetch(g + 16 + cu, 0, 1);
    }
#pragma unroll
    for (int i = 0; i < 8; ++i) {            // 2048 dwords of W tile
      int idx = tid + i * 256;
      int row = idx >> 4, cu = idx & 15;
      sB[idx] = ((const uint*)(W + (bn + row) * (long)K + k0))[cu];
    }
    __syncthreads();

    v16h af[2], bf[4];
#pragma unroll
    for (int ti = 0; ti < 2; ++ti)
      af[ti] = frag_a(As, 32, wm * 32 + ti * 16 + lane15, 0, laneHi);
#pragma unroll
    for (int tj = 0; tj < 4; ++tj)
      bf[tj] = frag_b(Bs, 32, wn * 64 + tj * 16 + lane15, 0, laneHi);
#pragma unroll
    for (int ti = 0; ti < 2; ++ti)
#pragma unroll
      for (int tj = 0; tj < 4; ++tj)
        acc[ti][tj] = wmma_f16(af[ti], bf[tj], acc[ti][tj]);
  }

#pragma unroll
  for (int ti = 0; ti < 2; ++ti)
#pragma unroll
    for (int tj = 0; tj < 4; ++tj)
#pragma unroll
      for (int r = 0; r < 8; ++r) {
        long row = bm + wm * 32 + ti * 16 + r + 8 * laneHi;   // C: M=r+8*hi
        long col = bn + wn * 64 + tj * 16 + lane15;           // C: N=lane&15
        float val = acc[ti][tj][r] + bias[col];
        if (MODE == 0) {
          int which = (int)(col / Dm);
          int d = (int)(col % Dm);
          int hh = d >> 6, hd = d & 63;
          long b = row / Sq, s = row % Sq;
          long off = ((b * Hh + hh) * Sq + s) * HDd + hd;
          if (which == 0)      qf[off] = (_Float16)(val * 0.125f); // HD^-0.5
          else if (which == 1) kf[off] = (_Float16)val;
          else                 vf[off] = (_Float16)val;
        } else {
          outf[row * (long)N + col] = val;
        }
      }
}

// ---------------------------------------------------------------- flash attn
// grid (S/128, H, B); block = 256 threads = 8 waves; each wave owns 16 q rows.
// Transposed formulation: S^T = K @ Q^T so softmax stats are per-lane scalars;
// O^T = V^T @ P^T accumulated in D layout (M=dim, N=qrow).
__launch_bounds__(256)
__global__ void flash_attn_kernel(const _Float16* __restrict__ Q,
                                  const _Float16* __restrict__ Kb,
                                  const _Float16* __restrict__ Vb,
                                  _Float16* __restrict__ Z) {
  __shared__ __align__(16) _Float16 Ks[32 * 64];     // [key][dim]
  __shared__ __align__(16) _Float16 Vt[64 * 32];     // [dim][key] (transposed)
  __shared__ __align__(16) _Float16 Ps[8][16 * 32];  // [wave][qrow][key]
  const int tid = threadIdx.x;
  const int lane = tid & 31, laneHi = lane >> 4, lane15 = lane & 15;
  const int wave = tid >> 5;
  const int qblk = blockIdx.x, hh = blockIdx.y, bb = blockIdx.z;
  const long bh = ((long)bb * Hh + hh) * Sq;       // row base in [B*H*S, 64]
  const int q0 = qblk * 128 + wave * 16;

  // Q as B-fragments (B[k=dim][n=qrow] = Q[qrow][dim]): two 16B chunks/lane
  v16h qb[2];
#pragma unroll
  for (int c = 0; c < 2; ++c)
    qb[c] = frag_b(Q + (bh + q0) * HDd, HDd, lane15, 32 * c, laneHi);

  float mcur = -3.0e38f, lcur = 0.f;   // per-lane stats for qrow = lane&15
  v8f oT[4] = {};                      // O^T: M=dim (4 tiles), N=qrow

  for (int j0 = 0; j0 < Sq; j0 += 32) {
    __syncthreads();
#pragma unroll
    for (int i = 0; i < 4; ++i) {      // stage 32x64 K and V tiles (1024 dw)
      int idx = tid + i * 256;
      int row = idx >> 5, cu = idx & 31;
      ((uint*)Ks)[idx] = ((const uint*)(Kb + (bh + j0 + row) * HDd))[cu];
      U32H2 u;
      u.u = ((const uint*)(Vb + (bh + j0 + row) * HDd))[cu];
      int d = cu * 2;
      Vt[d * 32 + row]       = u.h[0]; // transpose V into [dim][key]
      Vt[(d + 1) * 32 + row] = u.h[1];
    }
    __syncthreads();

    // S^T tiles: M = key (two 16-key tiles), N = qrow
    v8f st[2] = {};
#pragma unroll
    for (int kt = 0; kt < 2; ++kt)
#pragma unroll
      for (int c = 0; c < 2; ++c) {
        v16h kf = frag_a(Ks, HDd, kt * 16 + lane15, 32 * c, laneHi);
        st[kt] = wmma_f16(kf, qb[c], st[kt]);
      }

    // online softmax: in-lane reduce over 16 keys + one xor16 swap
    float vmax = st[0][0];
#pragma unroll
    for (int r = 1; r < 8; ++r) vmax = fmaxf(vmax, st[0][r]);
#pragma unroll
    for (int r = 0; r < 8; ++r) vmax = fmaxf(vmax, st[1][r]);
    vmax = fmaxf(vmax, xor16(vmax));
    float mnew = fmaxf(mcur, vmax);
    float rs = 0.f;
#pragma unroll
    for (int kt = 0; kt < 2; ++kt)
#pragma unroll
      for (int r = 0; r < 8; ++r) {
        float p = __expf(st[kt][r] - mnew);
        st[kt][r] = p;
        rs += p;
      }
    rs += xor16(rs);
    float scale = __expf(mcur - mnew);
    lcur = lcur * scale + rs;
    mcur = mnew;
#pragma unroll
    for (int dt = 0; dt < 4; ++dt)
#pragma unroll
      for (int r = 0; r < 8; ++r) oT[dt][r] *= scale;

    // repack P^T (D layout) -> B-fragment layout: Ps[qrow][key], 2x b128 each
#pragma unroll
    for (int kt = 0; kt < 2; ++kt) {
      v8h ph;
#pragma unroll
      for (int r = 0; r < 8; ++r) ph[r] = (_Float16)st[kt][r];
      *(v8h*)(&Ps[wave][lane15 * 32 + kt * 16 + 8 * laneHi]) = ph;
    }
    v16h pb = frag_b(Ps[wave], 32, lane15, 0, laneHi);

    // O^T += V^T @ P^T  (A: m=dim from transposed V tile, k=key)
#pragma unroll
    for (int dt = 0; dt < 4; ++dt) {
      v16h vf = frag_a(Vt, 32, dt * 16 + lane15, 0, laneHi);
      oT[dt] = wmma_f16(vf, pb, oT[dt]);
    }
  }

  // epilogue: lane owns one z row (q0+lane15); 4x contiguous 16B stores
  float inv = lcur > 0.f ? 1.0f / lcur : 0.f;
  long zrow = (long)bb * Sq + qblk * 128 + wave * 16 + lane15;
#pragma unroll
  for (int dt = 0; dt < 4; ++dt) {
    v8h zo;
#pragma unroll
    for (int r = 0; r < 8; ++r) zo[r] = (_Float16)(oT[dt][r] * inv);
    *(v8h*)(Z + zrow * Dm + hh * 64 + dt * 16 + 8 * laneHi) = zo;
  }
}

// ---------------------------------------------------------------- launch
extern "C" void kernel_launch(void* const* d_in, const int* in_sizes, int n_in,
                              void* d_out, int out_size, void* d_ws,
                              size_t ws_size, hipStream_t stream) {
  const float* x      = (const float*)d_in[0];
  const float* w_qkv  = (const float*)d_in[1];
  const float* b_qkv  = (const float*)d_in[2];
  const float* w_proj = (const float*)d_in[3];
  const float* b_proj = (const float*)d_in[4];
  float* out = (float*)d_out;

  const long MBS = (long)Bsz * Sq;  // 16384
  char* ws = (char*)d_ws;
  size_t off = 0;
  auto alloc_h = [&](long nelem) -> _Float16* {
    _Float16* p = (_Float16*)(ws + off);
    off += (size_t)nelem * sizeof(_Float16);
    off = (off + 255) & ~(size_t)255;
    return p;
  };
  _Float16* xh  = alloc_h(MBS * Dm);
  _Float16* wqh = alloc_h((long)3 * Dm * Dm);
  _Float16* wph = alloc_h((long)Dm * Dm);
  _Float16* qh  = alloc_h((long)Bsz * Hh * Sq * HDd);
  _Float16* kh  = alloc_h((long)Bsz * Hh * Sq * HDd);
  _Float16* vh  = alloc_h((long)Bsz * Hh * Sq * HDd);
  _Float16* zh  = alloc_h(MBS * Dm);

  {
    long n = MBS * Dm;  // 12582912
    cvt_f32_f16_kernel<<<(int)(n / 1024), 256, 0, stream>>>(x, xh, n);
  }
  {
    long n = (long)3 * Dm * Dm;  // 1769472
    cvt_f32_f16_kernel<<<(int)(n / 1024), 256, 0, stream>>>(w_qkv, wqh, n);
  }
  {
    long n = (long)Dm * Dm;  // 589824
    cvt_f32_f16_kernel<<<(int)(n / 1024), 256, 0, stream>>>(w_proj, wph, n);
  }

  // QKV projection: [16384 x 768] @ [768 x 2304] -> scatter q/k/v
  {
    dim3 grid(MBS / 128, (3 * Dm) / 128);  // (128, 18)
    gemm_wmma_kernel<0><<<grid, 256, 0, stream>>>(xh, wqh, b_qkv, 3 * Dm, Dm,
                                                  qh, kh, vh, nullptr);
  }

  // Flash attention -> z [16384 x 768] f16
  {
    dim3 grid(Sq / 128, Hh, Bsz);  // (8, 12, 16)
    flash_attn_kernel<<<grid, 256, 0, stream>>>(qh, kh, vh, zh);
  }

  // Output projection: [16384 x 768] @ [768 x 768] + b_proj -> f32 out
  {
    dim3 grid(MBS / 128, Dm / 128);  // (128, 6)
    gemm_wmma_kernel<1><<<grid, 256, 0, stream>>>(zh, wph, b_proj, Dm, Dm,
                                                  nullptr, nullptr, nullptr,
                                                  out);
  }
}
